// Glm4vVisionAttention_65987877536263
// MI455X (gfx1250) — matrix operand order repaired and checked by reading, not compiled, not run on hardware
//
#include <hip/hip_runtime.h>
#include <hip/hip_bf16.h>
#include <stdint.h>

// Problem constants (from reference)
#define S_TOK   3072
#define DMODEL  1536
#define NHEAD   12
#define HDIM    128
#define NSEQ    4
#define MT_S    (S_TOK / 16)             // 192 row tiles
#define SOFT_SCALE 0.08838834764831845f  // 1/sqrt(128)

#define LROW_K 136   // padded LDS row stride for K tile (halfs)
#define LROW_V 40    // padded LDS row stride for V tile (halfs)

typedef __attribute__((ext_vector_type(16))) __bf16       bf16x16;
typedef __attribute__((ext_vector_type(8)))  float        f32x8;
typedef __attribute__((ext_vector_type(4)))  unsigned int u32x4;
typedef __attribute__((ext_vector_type(4)))  int          i32x4;

union Frag16 { bf16x16 v; u32x4 q[2]; unsigned short u[16]; };
union FragC  { f32x8   v; float f[8]; };

// ---- gfx1250 async global->LDS path (guarded; falls back to sync copy) ----
#if defined(__has_builtin)
#  if __has_builtin(__builtin_amdgcn_global_load_async_to_lds_b128)
#    define HAVE_ASYNC_LDS 1
#  endif
#  if __has_builtin(__builtin_amdgcn_s_wait_asynccnt)
#    define HAVE_ASYNC_WAIT 1
#  endif
#endif
#ifndef HAVE_ASYNC_LDS
#  define HAVE_ASYNC_LDS 0
#endif
#ifndef HAVE_ASYNC_WAIT
#  define HAVE_ASYNC_WAIT 0
#endif

#if HAVE_ASYNC_LDS
typedef __attribute__((address_space(1))) i32x4 glob_i32x4;
typedef __attribute__((address_space(3))) i32x4 lds_i32x4;
#endif

__device__ __forceinline__ void async_copy16(const unsigned short* g, unsigned short* l) {
#if HAVE_ASYNC_LDS
  __builtin_amdgcn_global_load_async_to_lds_b128(
      (glob_i32x4*)const_cast<unsigned short*>(g), (lds_i32x4*)l, 0, 0);
#else
  *reinterpret_cast<u32x4*>(l) = *reinterpret_cast<const u32x4*>(g);
#endif
}

#if HAVE_ASYNC_LDS && HAVE_ASYNC_WAIT
#  define WAIT_ASYNC(N) __builtin_amdgcn_s_wait_asynccnt(N)
#else
#  define WAIT_ASYNC(N)
#endif

__device__ __forceinline__ unsigned short f2bf(float f) {
  unsigned int u = __float_as_uint(f);
  u += 0x7FFFu + ((u >> 16) & 1u);   // round-to-nearest-even
  return (unsigned short)(u >> 16);
}

// A-fragment (16x32 bf16, MxK): lane<16 -> row=lane, K {k0..+7, k0+16..+23}
//                              lane>=16 -> row=lane-16, K {k0+8..+15, k0+24..+31}
__device__ __forceinline__ bf16x16 load_frag_a(const unsigned short* __restrict__ base,
                                               int ld, int row0, int k0, int lane) {
  int r   = row0 + (lane & 15);
  int kof = k0 + ((lane >> 4) << 3);
  const unsigned short* p = base + (size_t)r * ld + kof;
  Frag16 f;
  f.q[0] = *reinterpret_cast<const u32x4*>(p);
  f.q[1] = *reinterpret_cast<const u32x4*>(p + 16);
  return f.v;
}

// B-fragment (32x16 bf16, KxN) from Bt[n][k] row-major:
// lane<16 -> col=lane, K {k0..+7, k0+16..+23} ; lane>=16 -> col=lane-16, K {k0+8..+15, k0+24..+31}
__device__ __forceinline__ bf16x16 load_frag_b(const unsigned short* __restrict__ base,
                                               int ld, int n0, int k0, int lane) {
  int n   = n0 + (lane & 15);
  int kof = k0 + ((lane >> 4) << 3);
  const unsigned short* p = base + (size_t)n * ld + kof;
  Frag16 f;
  f.q[0] = *reinterpret_cast<const u32x4*>(p);
  f.q[1] = *reinterpret_cast<const u32x4*>(p + 16);
  return f.v;
}

// Same B-fragment pattern but reading from an LDS tile (n relative to tile)
__device__ __forceinline__ bf16x16 frag_b_lds(const unsigned short* base,
                                              int ld, int k0, int lane) {
  int n   = lane & 15;
  int kof = k0 + ((lane >> 4) << 3);
  const unsigned short* p = base + n * ld + kof;
  Frag16 f;
  f.q[0] = *reinterpret_cast<const u32x4*>(p);
  f.q[1] = *reinterpret_cast<const u32x4*>(p + 16);
  return f.v;
}

__device__ __forceinline__ f32x8 wmma_bf16(bf16x16 a, bf16x16 b, f32x8 c) {
  return __builtin_amdgcn_wmma_f32_16x16x32_bf16(false, a, false, b, (short)0, c,
                                                 false, false);
}

// ---------------------------------------------------------------------------
// K1: fp32 -> bf16 conversion of hidden_states, qkv_w, proj_w
// ---------------------------------------------------------------------------
__global__ __launch_bounds__(256) void convert_bf16_kernel(
    const float* __restrict__ hs, const float* __restrict__ qkvw,
    const float* __restrict__ projw,
    unsigned short* __restrict__ hs_bf, unsigned short* __restrict__ qkvw_bf,
    unsigned short* __restrict__ projw_bf) {
  long long i = (long long)blockIdx.x * blockDim.x + threadIdx.x;
  const long long n_hs = (long long)S_TOK * DMODEL;
  const long long n_qw = 3LL * DMODEL * DMODEL;
  const long long n_pw = (long long)DMODEL * DMODEL;
  if (i < n_hs) hs_bf[i]    = f2bf(hs[i]);
  if (i < n_qw) qkvw_bf[i]  = f2bf(qkvw[i]);
  if (i < n_pw) projw_bf[i] = f2bf(projw[i]);
}

// RoPE + store for one (d, d+64) column-tile pair
__device__ __forceinline__ void rope_pair_store(
    unsigned short* __restrict__ dst, f32x8 accL, f32x8 accH,
    const float* __restrict__ qkv_b, const float* __restrict__ cosb,
    const float* __restrict__ sinb,
    int n_lo, int hh, int d_lo, int rbase, int col) {
  FragC cl, ch; cl.v = accL; ch.v = accH;
  float bias0 = qkv_b[n_lo + col];
  float bias1 = qkv_b[n_lo + 64 + col];
  int d_hi = d_lo + 64;
  for (int v = 0; v < 8; ++v) {
    int row = rbase + v;
    float lo = cl.f[v] + bias0;
    float hi = ch.f[v] + bias1;
    float cL = cosb[row * HDIM + d_lo], sL = sinb[row * HDIM + d_lo];
    float cH = cosb[row * HDIM + d_hi], sH = sinb[row * HDIM + d_hi];
    dst[(size_t)row * DMODEL + hh * HDIM + d_lo] = f2bf(lo * cL - hi * sL);
    dst[(size_t)row * DMODEL + hh * HDIM + d_hi] = f2bf(hi * cH + lo * sH);
  }
}

// ---------------------------------------------------------------------------
// K2: QKV GEMM (bf16 WMMA, fp32 acc) fused with bias + RoPE.
//   Each wave computes a 16x64 strip (4 tiles) so one A-frag feeds 4 WMMAs.
// ---------------------------------------------------------------------------
__global__ __launch_bounds__(256) void qkv_rope_kernel(
    const unsigned short* __restrict__ hs_bf,
    const unsigned short* __restrict__ qkvw_bf,
    const float* __restrict__ qkv_b,
    const float* __restrict__ cosb, const float* __restrict__ sinb,
    unsigned short* __restrict__ q_bf, unsigned short* __restrict__ k_bf,
    unsigned short* __restrict__ v_t) {
  int lane  = threadIdx.x & 31;
  int wave  = blockIdx.x * 8 + (threadIdx.x >> 5);
  int mtile = wave % MT_S;
  int t     = wave / MT_S;          // 0..71
  int m0    = mtile * 16;
  int col   = lane & 15;
  int rbase = m0 + ((lane >> 4) << 3);

  if (t < 48) {                     // ---- q or k : two RoPE pairs ----
    int qk  = t / 24;               // 0 = q, 1 = k
    int tt  = t % 24;
    int hh  = tt / 2;
    int dt2 = tt % 2;               // which 32-wide d slice (d < 64)
    int nL  = qk * DMODEL + hh * HDIM + dt2 * 32;
    f32x8 aL0 = {}, aL1 = {}, aH0 = {}, aH1 = {};
    for (int k0 = 0; k0 < DMODEL; k0 += 32) {
      bf16x16 a   = load_frag_a(hs_bf, DMODEL, m0, k0, lane);
      bf16x16 bl0 = load_frag_b(qkvw_bf, DMODEL, nL,      k0, lane);
      bf16x16 bl1 = load_frag_b(qkvw_bf, DMODEL, nL + 16, k0, lane);
      bf16x16 bh0 = load_frag_b(qkvw_bf, DMODEL, nL + 64, k0, lane);
      bf16x16 bh1 = load_frag_b(qkvw_bf, DMODEL, nL + 80, k0, lane);
      aL0 = wmma_bf16(a, bl0, aL0);
      aL1 = wmma_bf16(a, bl1, aL1);
      aH0 = wmma_bf16(a, bh0, aH0);
      aH1 = wmma_bf16(a, bh1, aH1);
    }
    unsigned short* dst = (qk == 0) ? q_bf : k_bf;
    rope_pair_store(dst, aL0, aH0, qkv_b, cosb, sinb, nL,      hh,
                    dt2 * 32 + col,      rbase, col);
    rope_pair_store(dst, aL1, aH1, qkv_b, cosb, sinb, nL + 16, hh,
                    dt2 * 32 + 16 + col, rbase, col);
  } else {                          // ---- v strip, store transposed ----
    int t2  = t - 48;
    int hh  = t2 / 2;
    int dt4 = t2 % 2;               // which 64-wide d slice
    int n0  = 2 * DMODEL + hh * HDIM + dt4 * 64;
    f32x8 acc[4];
    for (int j = 0; j < 4; ++j) { f32x8 z = {}; acc[j] = z; }
    for (int k0 = 0; k0 < DMODEL; k0 += 32) {
      bf16x16 a = load_frag_a(hs_bf, DMODEL, m0, k0, lane);
      for (int j = 0; j < 4; ++j) {
        bf16x16 b = load_frag_b(qkvw_bf, DMODEL, n0 + j * 16, k0, lane);
        acc[j] = wmma_bf16(a, b, acc[j]);
      }
    }
    for (int j = 0; j < 4; ++j) {
      FragC c; c.v = acc[j];
      float bias = qkv_b[n0 + j * 16 + col];
      int d = dt4 * 64 + j * 16 + col;
      Frag16 pk;
      for (int v = 0; v < 8; ++v) pk.u[v] = f2bf(c.f[v] + bias);
      size_t idx = ((size_t)hh * HDIM + d) * S_TOK + rbase;   // 8 consecutive s
      *reinterpret_cast<u32x4*>(v_t + idx) = pk.q[0];
    }
  }
}

// ---------------------------------------------------------------------------
// K3: flash attention. 8 waves/block share one (head, seq-block) and stage
//   32-key K/V tiles in LDS via async global->LDS loads (double-buffered).
// ---------------------------------------------------------------------------
__global__ __launch_bounds__(256) void attn_kernel(
    const unsigned short* __restrict__ q_bf,
    const unsigned short* __restrict__ k_bf,
    const unsigned short* __restrict__ v_t,
    const int* __restrict__ cu,
    unsigned short* __restrict__ attn_bf) {
  __shared__ __align__(16) unsigned short k_lds[2][32 * LROW_K];   // 2 x 8704 B
  __shared__ __align__(16) unsigned short v_lds[2][128 * LROW_V];  // 2 x 10240 B
  __shared__ __align__(16) unsigned short p_lds[8][16 * 32];       // 8 KB

  int tid   = threadIdx.x;
  int lane  = tid & 31;
  int wslot = tid >> 5;
  int b     = blockIdx.x;
  int mg    = b % 6;
  int rest  = b / 6;
  int blk   = rest % NSEQ;
  int hh    = rest / NSEQ;
  int s_start = cu[blk];
  int s_end   = cu[blk + 1];
  int m0 = s_start + (mg * 8 + wslot) * 16;

  const unsigned short* kg = k_bf + hh * HDIM;                  // + key*DMODEL
  const unsigned short* vg = v_t + (size_t)hh * HDIM * S_TOK;   // + d*S_TOK

  // Preload the 4 Q A-fragments (K = 128)
  bf16x16 aq[4];
  for (int i = 0; i < 4; ++i)
    aq[i] = load_frag_a(q_bf + hh * HDIM, DMODEL, m0, i * 32, lane);

  f32x8 acc_o[8];
  for (int i = 0; i < 8; ++i) { f32x8 z = {}; acc_o[i] = z; }
  float mrun[8], lrun[8];
  for (int v = 0; v < 8; ++v) { mrun[v] = -1.0e30f; lrun[v] = 0.0f; }

  int col = lane & 15;
  int rb  = (lane >> 4) << 3;
  unsigned short* pl = p_lds[wslot];

  // Stage K tile (32 keys x 128 dims) and V tile (128 dims x 32 keys):
  // 512 + 512 b128 chunks, 4 async issues per thread.
  auto fill_kv = [&](int buf, int kk) {
    for (int c = tid; c < 512; c += 256) {
      int row = c >> 4, cc = c & 15;
      async_copy16(kg + (size_t)(kk + row) * DMODEL + cc * 8,
                   &k_lds[buf][row * LROW_K + cc * 8]);
    }
    for (int c = tid; c < 512; c += 256) {
      int d = c >> 2, cc = c & 3;
      async_copy16(vg + (size_t)d * S_TOK + kk + cc * 8,
                   &v_lds[buf][d * LROW_V + cc * 8]);
    }
  };

  int nsteps = (s_end - s_start) >> 5;   // 24 for 768-length blocks
  fill_kv(0, s_start);

  for (int it = 0; it < nsteps; ++it) {
    int kk  = s_start + it * 32;
    int cur = it & 1;
    __syncthreads();                     // buffer cur^1 free for refill
    if (it + 1 < nsteps) {
      fill_kv(cur ^ 1, kk + 32);
      WAIT_ASYNC(4);                     // oldest 4 (buffer cur) complete
    } else {
      WAIT_ASYNC(0);
    }
    __syncthreads();                     // all waves' fills of cur visible

    // ---- scores: two 16x16 tiles over K=128, B-frags from LDS ----
    const unsigned short* kb = &k_lds[cur][0];
    f32x8 s0 = {}, s1 = {};
    for (int i = 0; i < 4; ++i) {
      bf16x16 b0 = frag_b_lds(kb,               LROW_K, i * 32, lane);
      bf16x16 b1 = frag_b_lds(kb + 16 * LROW_K, LROW_K, i * 32, lane);
      s0 = wmma_bf16(aq[i], b0, s0);
      s1 = wmma_bf16(aq[i], b1, s1);
    }
    FragC t0, t1; t0.v = s0; t1.v = s1;

    // ---- online softmax update (row reductions across 16 lanes) ----
    float corrv[8];
    for (int v = 0; v < 8; ++v) {
      t0.f[v] *= SOFT_SCALE; t1.f[v] *= SOFT_SCALE;
      float x = fmaxf(t0.f[v], t1.f[v]);
      for (int m = 8; m >= 1; m >>= 1) x = fmaxf(x, __shfl_xor(x, m, 32));
      float mnew = fmaxf(mrun[v], x);
      float corr = __expf(mrun[v] - mnew);
      mrun[v] = mnew;
      float p0 = __expf(t0.f[v] - mnew);
      float p1 = __expf(t1.f[v] - mnew);
      float ps = p0 + p1;
      for (int m = 8; m >= 1; m >>= 1) ps += __shfl_xor(ps, m, 32);
      lrun[v] = lrun[v] * corr + ps;
      t0.f[v] = p0; t1.f[v] = p1;
      corrv[v] = corr;
    }
    for (int dt = 0; dt < 8; ++dt) {
      FragC o; o.v = acc_o[dt];
      for (int v = 0; v < 8; ++v) o.f[v] *= corrv[v];
      acc_o[dt] = o.v;
    }

    // ---- C-layout -> A-layout via per-wave LDS slot ----
    for (int v = 0; v < 8; ++v) {
      int row = rb + v;
      pl[row * 32 + col]      = f2bf(t0.f[v]);
      pl[row * 32 + 16 + col] = f2bf(t1.f[v]);
    }
    Frag16 pa;
    {
      int r   = lane & 15;
      int kof = (lane >> 4) << 3;
      const unsigned short* src = pl + r * 32 + kof;
      pa.q[0] = *reinterpret_cast<const u32x4*>(src);
      pa.q[1] = *reinterpret_cast<const u32x4*>(src + 16);
    }

    // ---- PV: 8 d-tiles, K=32, B-frags from LDS V tile ----
    const unsigned short* vb = &v_lds[cur][0];
    for (int dt = 0; dt < 8; ++dt) {
      bf16x16 bv = frag_b_lds(vb + dt * 16 * LROW_V, LROW_V, 0, lane);
      acc_o[dt] = wmma_bf16(pa.v, bv, acc_o[dt]);
    }
  }

  // ---- finalize: divide by row sum, write bf16 attn[s][h*128+d] ----
  int rowb = m0 + rb;
  for (int dt = 0; dt < 8; ++dt) {
    FragC o; o.v = acc_o[dt];
    for (int v = 0; v < 8; ++v) {
      float val = o.f[v] / lrun[v];
      attn_bf[(size_t)(rowb + v) * DMODEL + hh * HDIM + dt * 16 + col] = f2bf(val);
    }
  }
}

// ---------------------------------------------------------------------------
// K4: output projection (attn @ proj_w.T), 16x64 strip per wave, fp32 result
// ---------------------------------------------------------------------------
__global__ __launch_bounds__(256) void proj_kernel(
    const unsigned short* __restrict__ attn_bf,
    const unsigned short* __restrict__ projw_bf,
    float* __restrict__ out) {
  int lane  = threadIdx.x & 31;
  int wave  = blockIdx.x * 8 + (threadIdx.x >> 5);
  int mtile = wave % MT_S;
  int ng    = wave / MT_S;           // 0..23
  int m0 = mtile * 16, n0 = ng * 64;
  f32x8 acc[4];
  for (int j = 0; j < 4; ++j) { f32x8 z = {}; acc[j] = z; }
  for (int k0 = 0; k0 < DMODEL; k0 += 32) {
    bf16x16 a = load_frag_a(attn_bf, DMODEL, m0, k0, lane);
    for (int j = 0; j < 4; ++j) {
      bf16x16 b = load_frag_b(projw_bf, DMODEL, n0 + j * 16, k0, lane);
      acc[j] = wmma_bf16(a, b, acc[j]);
    }
  }
  int col = lane & 15;
  int rb  = m0 + ((lane >> 4) << 3);
  for (int j = 0; j < 4; ++j) {
    FragC c; c.v = acc[j];
    for (int v = 0; v < 8; ++v)
      out[(size_t)(rb + v) * DMODEL + n0 + j * 16 + col] = c.f[v];
  }
}

// ---------------------------------------------------------------------------
extern "C" void kernel_launch(void* const* d_in, const int* in_sizes, int n_in,
                              void* d_out, int out_size, void* d_ws, size_t ws_size,
                              hipStream_t stream) {
  const float* hs    = (const float*)d_in[0];
  const float* cosb  = (const float*)d_in[1];
  const float* sinb  = (const float*)d_in[2];
  const float* qkvw  = (const float*)d_in[3];
  const float* qkvb  = (const float*)d_in[4];
  const float* projw = (const float*)d_in[5];
  const int*   cu    = (const int*)d_in[6];
  float* out = (float*)d_out;

  // Workspace layout (bf16 buffers), ~63 MB total
  char* ws = (char*)d_ws;
  size_t off = 0;
  auto take = [&](size_t bytes) {
    void* p = ws + off;
    off += (bytes + 255) & ~(size_t)255;
    return p;
  };
  unsigned short* hs_bf    = (unsigned short*)take((size_t)S_TOK * DMODEL * 2);
  unsigned short* qkvw_bf  = (unsigned short*)take(3ull * DMODEL * DMODEL * 2);
  unsigned short* projw_bf = (unsigned short*)take((size_t)DMODEL * DMODEL * 2);
  unsigned short* q_bf     = (unsigned short*)take((size_t)S_TOK * DMODEL * 2);
  unsigned short* k_bf     = (unsigned short*)take((size_t)S_TOK * DMODEL * 2);
  unsigned short* v_t      = (unsigned short*)take((size_t)S_TOK * DMODEL * 2);
  unsigned short* attn_bf  = (unsigned short*)take((size_t)S_TOK * DMODEL * 2);
  (void)ws_size; (void)in_sizes; (void)n_in; (void)out_size;

  // K1: convert fp32 -> bf16 (largest array is qkv_w: 3*D*D)
  long long nconv = 3LL * DMODEL * DMODEL;
  int cblocks = (int)((nconv + 255) / 256);
  convert_bf16_kernel<<<cblocks, 256, 0, stream>>>(hs, qkvw, projw,
                                                   hs_bf, qkvw_bf, projw_bf);

  // K2: QKV + bias + RoPE. 72 strip tasks * 192 row tiles = 13824 waves
  qkv_rope_kernel<<<13824 / 8, 256, 0, stream>>>(hs_bf, qkvw_bf, qkvb, cosb, sinb,
                                                 q_bf, k_bf, v_t);

  // K3: flash attention. 12 heads * 4 blocks * 6 mtile groups = 288 blocks
  attn_kernel<<<288, 256, 0, stream>>>(q_bf, k_bf, v_t, cu, attn_bf);

  // K4: projection. 24 strip tasks * 192 row tiles = 4608 waves
  proj_kernel<<<4608 / 8, 256, 0, stream>>>(attn_bf, projw_bf, out);
}
